// LoraModel_84550726189634
// MI455X (gfx1250) — compile-verified, matrix operand order checked by reading, not run on hardware
//
#include <hip/hip_runtime.h>
#include <hip/hip_bf16.h>

// ---------------------------------------------------------------------------
// LoRA transformer forward for MI455X (gfx1250, wave32, WMMA).
//   L=8, D=1024, H=16, hd=64, R=16, B=2, S=1024.  ~206 GFLOP vs ~0.5 GB of
// traffic @ 23.3 TB/s -> compute bound -> everything on v_wmma_f32_16x16x32_f16
// (f16 operands, fp32 accumulation). Weights are merged AND transposed once per
// layer so both GEMM operands stage into LDS as linear 16B copies, enabling the
// CDNA5 async global->LDS path (GLOBAL_LOAD_ASYNC_TO_LDS_B128, ASYNCcnt).
// ---------------------------------------------------------------------------

typedef _Float16 v16h __attribute__((ext_vector_type(16)));
typedef _Float16 v8h  __attribute__((ext_vector_type(8)));
typedef float    v8f  __attribute__((ext_vector_type(8)));
typedef int      v4i  __attribute__((ext_vector_type(4)));

typedef __attribute__((address_space(1))) v4i gv4i;   // global int4
typedef __attribute__((address_space(3))) v4i lv4i;   // LDS int4

union H16 { v16h v; v8h h[2]; };

#if defined(__HIP_DEVICE_COMPILE__) && __has_builtin(__builtin_amdgcn_global_load_async_to_lds_b128)
#define HAVE_ASYNC_LDS 1
#else
#define HAVE_ASYNC_LDS 0
#endif

__device__ __forceinline__ v8f wmma16x16x32(v16h a, v16h b, v8f c) {
    return __builtin_amdgcn_wmma_f32_16x16x32_f16(false, a, false, b,
                                                  (short)0, c, false, false);
}

// 16-byte global -> LDS copy (async when available, else through VGPRs).
__device__ __forceinline__ void copy_b128(const _Float16* g, _Float16* l) {
#if HAVE_ASYNC_LDS
    __builtin_amdgcn_global_load_async_to_lds_b128(
        (gv4i*)(unsigned long long)(size_t)g,
        (lv4i*)(unsigned)(size_t)l, 0, 0);
#else
    *(v8h*)l = *(const v8h*)g;
#endif
}

__device__ __forceinline__ void async_fence() {
#if HAVE_ASYNC_LDS
#if __has_builtin(__builtin_amdgcn_s_wait_asynccnt)
    __builtin_amdgcn_s_wait_asynccnt(0);
#else
    asm volatile("s_wait_asynccnt 0x0" ::: "memory");
#endif
#endif
}

// fragment load from padded LDS row (CDNA5 A/B operand layout:
// lane<16 -> K{0..7,16..23}, lane>=16 -> K{8..15,24..31})
__device__ __forceinline__ v16h load_frag(const _Float16* row, int koff) {
    H16 u;
    u.h[0] = *(const v8h*)(row + koff);
    u.h[1] = *(const v8h*)(row + koff + 16);
    return u.v;
}

// ---------------------------------------------------------------------------
// Kernel 1: merged + transposed weight  Wt[n][k] = f16(W + A @ B)[k][n]
//   LDS-tiled 32x32 transpose, coalesced reads and writes. A:[1024,16], B:[16,N]
// ---------------------------------------------------------------------------
__global__ __launch_bounds__(256)
void merge_lora_t(const float* __restrict__ W, const float* __restrict__ A,
                  const float* __restrict__ Bm, _Float16* __restrict__ Wt, int N) {
    __shared__ _Float16 tile[32][33];
    const int tx = threadIdx.x, ty = threadIdx.y;
    const int i0 = blockIdx.y * 32;   // k rows of W
    const int j0 = blockIdx.x * 32;   // n cols of W
#pragma unroll
    for (int g = 0; g < 4; ++g) {
        const int i = i0 + ty + g * 8;
        const int j = j0 + tx;
        float acc = W[(size_t)i * N + j];
#pragma unroll
        for (int r = 0; r < 16; ++r)
            acc = fmaf(A[i * 16 + r], Bm[(size_t)r * N + j], acc);
        tile[ty + g * 8][tx] = (_Float16)acc;
    }
    __syncthreads();
#pragma unroll
    for (int g = 0; g < 4; ++g) {
        const int j = j0 + ty + g * 8;   // output row (n)
        const int i = i0 + tx;           // output col (k)
        Wt[(size_t)j * 1024 + i] = tile[tx][ty + g * 8];
    }
}

// ---------------------------------------------------------------------------
// Kernel 2: WMMA GEMM (TN)  C[M,N] = Ah[M,K] * Bt[N,K]^T (+ Res)
//   Block tile 128x128, BK=32, 256 threads = 8 waves (4x2), wave tile 32x64
//   -> 8 wmma / k-step. Double-buffered LDS, async global->LDS staging.
//   Always writes f16 OutH; optionally fp32 OutF (+residual).
// ---------------------------------------------------------------------------
template <bool ADD_RES, bool WRITE_F32>
__global__ __launch_bounds__(256, 1)
void gemm_wmma_tn(const _Float16* __restrict__ Ah, const _Float16* __restrict__ Bt,
                  const float* __restrict__ Res, float* __restrict__ OutF,
                  _Float16* __restrict__ OutH, int M, int N, int K) {
    __shared__ _Float16 As[2][128 * 40];
    __shared__ _Float16 Bs[2][128 * 40];

    const int tid  = threadIdx.x;
    const int wid  = tid >> 5;
    const int lane = tid & 31;
    const int l16  = lane & 15;
    const int koff = (lane < 16) ? 0 : 8;

    const int m0 = blockIdx.y * 128;
    const int n0 = blockIdx.x * 128;
    const int moff = (wid & 3) * 32;   // wave M sub-tile (2x16)
    const int noff = (wid >> 2) * 64;  // wave N sub-tile (4x16)

    const v8f zero = {0.f, 0.f, 0.f, 0.f, 0.f, 0.f, 0.f, 0.f};
    v8f acc[2][4] = {{zero, zero, zero, zero}, {zero, zero, zero, zero}};

    // staging: thread -> 32B of A and 32B of B per buffer
    const int sr = tid >> 1;            // 0..127
    const int sc = (tid & 1) * 16;      // 0 or 16

    const int NK = K >> 5;
    int buf = 0;

    // prologue: stage k-tile 0 into buffer 0
    {
        const _Float16* ga = Ah + (size_t)(m0 + sr) * K + sc;
        const _Float16* gb = Bt + (size_t)(n0 + sr) * K + sc;
        _Float16* la = &As[0][sr * 40 + sc];
        _Float16* lb = &Bs[0][sr * 40 + sc];
        copy_b128(ga, la);      copy_b128(ga + 8, la + 8);
        copy_b128(gb, lb);      copy_b128(gb + 8, lb + 8);
    }

    for (int kt = 0; kt < NK; ++kt) {
        async_fence();
        __syncthreads();
        if (kt + 1 < NK) {
            const int k0 = (kt + 1) << 5;
            const _Float16* ga = Ah + (size_t)(m0 + sr) * K + k0 + sc;
            const _Float16* gb = Bt + (size_t)(n0 + sr) * K + k0 + sc;
            _Float16* la = &As[buf ^ 1][sr * 40 + sc];
            _Float16* lb = &Bs[buf ^ 1][sr * 40 + sc];
            copy_b128(ga, la);  copy_b128(ga + 8, la + 8);
            copy_b128(gb, lb);  copy_b128(gb + 8, lb + 8);
        }

        v16h af[2], bf[4];
#pragma unroll
        for (int mi = 0; mi < 2; ++mi)
            af[mi] = load_frag(&As[buf][(moff + mi * 16 + l16) * 40], koff);
#pragma unroll
        for (int ni = 0; ni < 4; ++ni)
            bf[ni] = load_frag(&Bs[buf][(noff + ni * 16 + l16) * 40], koff);
#pragma unroll
        for (int mi = 0; mi < 2; ++mi)
#pragma unroll
            for (int ni = 0; ni < 4; ++ni)
                acc[mi][ni] = wmma16x16x32(af[mi], bf[ni], acc[mi][ni]);
        buf ^= 1;
    }

    // epilogue: C layout lane<16 -> M=v, lane>=16 -> M=v+8
#pragma unroll
    for (int mi = 0; mi < 2; ++mi) {
#pragma unroll
        for (int ni = 0; ni < 4; ++ni) {
#pragma unroll
            for (int v = 0; v < 8; ++v) {
                const int row = m0 + moff + mi * 16 + ((lane < 16) ? v : v + 8);
                const int col = n0 + noff + ni * 16 + l16;
                float x = acc[mi][ni][v];
                if (ADD_RES) x += Res[(size_t)row * N + col];
                if (WRITE_F32) OutF[(size_t)row * N + col] = x;
                OutH[(size_t)row * N + col] = (_Float16)x;
            }
        }
    }
}

// ---------------------------------------------------------------------------
// Kernel 3: causal flash attention over qkv16 [B*S, 3D] (f16), hd=64.
//   Grid (B*H, S/64); 128 threads = 4 waves; wave owns 16 query rows.
//   Q/K fragments load straight from global; V staged transposed [hd][key] in
//   LDS; P round-trips LDS for C->A layout. Online softmax fp32.
// ---------------------------------------------------------------------------
__global__ __launch_bounds__(128)
void attn_flash(const _Float16* __restrict__ qkv, _Float16* __restrict__ o16) {
    constexpr int S = 1024, D3 = 3072;
    __shared__ _Float16 Vt[64 * 40];
    __shared__ _Float16 Pl[4][16 * 40];

    const int tid  = threadIdx.x;
    const int wid  = tid >> 5;
    const int lane = tid & 31;
    const int l16  = lane & 15;
    const int koff = (lane < 16) ? 0 : 8;

    const int bh = blockIdx.x, b = bh >> 4, h = bh & 15;
    const int qbase = blockIdx.y * 64;

    const _Float16* base = qkv + (size_t)b * S * D3;
    const int qcol = h * 64, kcol = 1024 + h * 64, vcol = 2048 + h * 64;

    const int qrow = qbase + wid * 16 + l16;
    const _Float16* qp = base + (size_t)qrow * D3 + qcol;
    H16 aq0, aq1;
    aq0.h[0] = *(const v8h*)(qp + koff);      aq0.h[1] = *(const v8h*)(qp + koff + 16);
    aq1.h[0] = *(const v8h*)(qp + 32 + koff); aq1.h[1] = *(const v8h*)(qp + 32 + koff + 16);

    const v8f zero = {0.f, 0.f, 0.f, 0.f, 0.f, 0.f, 0.f, 0.f};
    v8f O[4] = {zero, zero, zero, zero};
    float mrow[8], lrow[8];
#pragma unroll
    for (int v = 0; v < 8; ++v) { mrow[v] = -1e30f; lrow[v] = 0.f; }

    const int nch = (qbase + 64) >> 5;
    for (int ch = 0; ch < nch; ++ch) {
        const int kb = ch * 32;
        __syncthreads();
        {   // stage V chunk transposed: Vt[hd][key]
            const int kr = tid >> 2, c8 = (tid & 3) * 8;
#pragma unroll
            for (int g = 0; g < 2; ++g) {
                const int cc = c8 + g * 32;
                v8h vv = *(const v8h*)(base + (size_t)(kb + kr) * D3 + vcol + cc);
#pragma unroll
                for (int i = 0; i < 8; ++i) Vt[(cc + i) * 40 + kr] = vv[i];
            }
        }
        __syncthreads();

        v8f st[2];
#pragma unroll
        for (int t = 0; t < 2; ++t) {
            const int key = kb + t * 16 + l16;
            const _Float16* kp = base + (size_t)key * D3 + kcol;
            H16 b0, b1;
            b0.h[0] = *(const v8h*)(kp + koff);      b0.h[1] = *(const v8h*)(kp + koff + 16);
            b1.h[0] = *(const v8h*)(kp + 32 + koff); b1.h[1] = *(const v8h*)(kp + 32 + koff + 16);
            v8f s = zero;
            s = wmma16x16x32(aq0.v, b0.v, s);
            s = wmma16x16x32(aq1.v, b1.v, s);
            st[t] = s;
        }

        const int qv0 = qbase + wid * 16;
#pragma unroll
        for (int v = 0; v < 8; ++v) {
            const int Mr = (lane < 16) ? v : v + 8;
            const int qi = qv0 + Mr;
            const int k0i = kb + l16, k1i = k0i + 16;
            float s0 = (k0i <= qi) ? st[0][v] * 0.125f : -1e30f;
            float s1 = (k1i <= qi) ? st[1][v] * 0.125f : -1e30f;
            float r = fmaxf(s0, s1);
#pragma unroll
            for (int off = 1; off < 16; off <<= 1)
                r = fmaxf(r, __shfl_xor(r, off, 32));
            const float mn = fmaxf(mrow[v], r);
            const float alpha = __expf(mrow[v] - mn);
            const float p0 = __expf(s0 - mn);
            const float p1 = __expf(s1 - mn);
            float rs = p0 + p1;
#pragma unroll
            for (int off = 1; off < 16; off <<= 1)
                rs += __shfl_xor(rs, off, 32);
            lrow[v] = lrow[v] * alpha + rs;
            mrow[v] = mn;
#pragma unroll
            for (int t = 0; t < 4; ++t) O[t][v] *= alpha;
            Pl[wid][Mr * 40 + l16]      = (_Float16)p0;
            Pl[wid][Mr * 40 + 16 + l16] = (_Float16)p1;
        }
        __syncthreads();

        const _Float16* prow = &Pl[wid][l16 * 40];
        H16 pf;
        pf.h[0] = *(const v8h*)(prow + koff);
        pf.h[1] = *(const v8h*)(prow + koff + 16);
#pragma unroll
        for (int t = 0; t < 4; ++t) {
            const _Float16* vrow = &Vt[(t * 16 + l16) * 40];
            H16 vf;
            vf.h[0] = *(const v8h*)(vrow + koff);
            vf.h[1] = *(const v8h*)(vrow + koff + 16);
            O[t] = wmma16x16x32(pf.v, vf.v, O[t]);
        }
    }

#pragma unroll
    for (int v = 0; v < 8; ++v) {
        const float inv = 1.0f / lrow[v];
        const int Mr = (lane < 16) ? v : v + 8;
        const int orow = b * S + qbase + wid * 16 + Mr;
#pragma unroll
        for (int t = 0; t < 4; ++t) {
            const int oc = h * 64 + t * 16 + l16;
            o16[(size_t)orow * 1024 + oc] = (_Float16)(O[t][v] * inv);
        }
    }
}

// ---------------------------------------------------------------------------
// Kernel 4: f32 -> f16 cast (seed h16 from x)
// ---------------------------------------------------------------------------
__global__ __launch_bounds__(256)
void cast_to_h16(const float* __restrict__ x, _Float16* __restrict__ y) {
    const int i = (blockIdx.x * 256 + threadIdx.x) * 4;
    float4 f = *(const float4*)(x + i);
    y[i + 0] = (_Float16)f.x; y[i + 1] = (_Float16)f.y;
    y[i + 2] = (_Float16)f.z; y[i + 3] = (_Float16)f.w;
}

// ---------------------------------------------------------------------------
// Host: per-layer pipeline. d_out doubles as the fp32 hidden state h; an f16
// mirror h16 is kept in ws so GEMM A-operands are always linear f16.
// Workspace (28 MB): Wqt[3072,1024] | Wpt[1024,1024] | qkv16[2048,3072] |
//                    o16[2048,1024] | h16[2048,1024]   (all f16)
// ---------------------------------------------------------------------------
extern "C" void kernel_launch(void* const* d_in, const int* in_sizes, int n_in,
                              void* d_out, int out_size, void* d_ws, size_t ws_size,
                              hipStream_t stream) {
    (void)in_sizes; (void)n_in; (void)out_size; (void)ws_size;
    constexpr int D = 1024, S = 1024, Bb = 2, D3 = 3072, L = 8;

    const float* x     = (const float*)d_in[0];
    const float* Wqkv  = (const float*)d_in[1];
    const float* Aqkv  = (const float*)d_in[2];
    const float* Bqkv  = (const float*)d_in[3];
    const float* Wproj = (const float*)d_in[4];
    const float* Aproj = (const float*)d_in[5];
    const float* Bproj = (const float*)d_in[6];

    float* h = (float*)d_out;   // running hidden state [2048, 1024] fp32

    char* ws = (char*)d_ws;
    size_t off = 0;
    _Float16* Wqt   = (_Float16*)(ws + off); off += (size_t)D3 * D * 2;
    _Float16* Wpt   = (_Float16*)(ws + off); off += (size_t)D * D * 2;
    _Float16* qkv16 = (_Float16*)(ws + off); off += (size_t)Bb * S * D3 * 2;
    _Float16* o16   = (_Float16*)(ws + off); off += (size_t)Bb * S * D * 2;
    _Float16* h16   = (_Float16*)(ws + off);

    const int M = Bb * S;  // 2048
    (void)hipMemcpyAsync(h, x, (size_t)M * D * sizeof(float),
                         hipMemcpyDeviceToDevice, stream);
    cast_to_h16<<<(M * D) / 1024, 256, 0, stream>>>(x, h16);

    for (int l = 0; l < L; ++l) {
        merge_lora_t<<<dim3(D3 / 32, D / 32), dim3(32, 8), 0, stream>>>(
            Wqkv + (size_t)l * D * D3, Aqkv + (size_t)l * D * 16,
            Bqkv + (size_t)l * 16 * D3, Wqt, D3);
        merge_lora_t<<<dim3(D / 32, D / 32), dim3(32, 8), 0, stream>>>(
            Wproj + (size_t)l * D * D, Aproj + (size_t)l * D * 16,
            Bproj + (size_t)l * 16 * D, Wpt, D);

        // qkv16 = h16 @ Wqt^T
        gemm_wmma_tn<false, false><<<dim3(D3 / 128, M / 128), 256, 0, stream>>>(
            h16, Wqt, nullptr, nullptr, qkv16, M, D3, D);

        // causal flash attention
        attn_flash<<<dim3(Bb * 16, S / 64), 128, 0, stream>>>(qkv16, o16);

        // h = h + o16 @ Wpt^T  (fp32 residual) and refresh h16 mirror
        gemm_wmma_tn<true, true><<<dim3(D / 128, M / 128), 256, 0, stream>>>(
            o16, Wpt, h, h, h16, M, D, D);
    }
}
